// MaskYolo_49847390437664
// MI455X (gfx1250) — compile-verified, matrix operand order checked by reading, not run on hardware
//
#include <hip/hip_runtime.h>
#include <hip/hip_bf16.h>

typedef __attribute__((ext_vector_type(16))) _Float16 v16h;
typedef __attribute__((ext_vector_type(8)))  _Float16 v8h;
typedef __attribute__((ext_vector_type(8)))  float    v8f;

// ---------------- constants ----------------
#define NB        8
#define NCAND     1360       // 32^2+16^2+8^2+4^2
#define KPRE      512
#define RLIM      100
#define NROI      (NB*RLIM)  // 800
#define CMERGE    480
#define KDIM      (CMERGE*9) // 4320
#define NCHUNK    15         // 480 / 32
#define MSZ       16
#define PATCHSZ   256
#define PADW      18         // 16 + halo
#define SHALVES   (PADW*PADW*32)

// workspace layout (bytes)
#define OFF_BOXES   ((size_t)0)                                  // 8*1360*4 f
#define OFF_SCORES  (OFF_BOXES  + (size_t)NB*NCAND*4*4)
#define OFF_LABELS  (OFF_SCORES + (size_t)NB*NCAND*4)
#define OFF_BSEL    (OFF_LABELS + (size_t)NB*NCAND*4)
#define OFF_SSEL    (OFF_BSEL   + (size_t)NROI*4*4)
#define OFF_LSEL    (OFF_SSEL   + (size_t)NROI*4)
#define OFF_VSEL    (OFF_LSEL   + (size_t)NROI*4)
#define OFF_PANEL   (((OFF_VSEL + (size_t)NROI*4 + 127)/128)*128) // 276480 halves
#define OFF_LM      (OFF_PANEL  + (size_t)KDIM*64*2)
#define OFF_CNT     (OFF_LM     + (size_t)NROI*256*4)

__device__ __forceinline__ float sigmf(float x){ return 1.f/(1.f+__expf(-x)); }

// ---------------- decode ----------------
__global__ __launch_bounds__(256) void decode_kernel(
    const float* __restrict__ p0, const float* __restrict__ p1,
    const float* __restrict__ p2, const float* __restrict__ p3,
    float* __restrict__ boxes, float* __restrict__ scores, int* __restrict__ labels)
{
  int id = blockIdx.x*256 + threadIdx.x;
  if (id >= NB*NCAND) return;
  int b = id / NCAND, n = id % NCAND;
  const float* p; int g, s, off;
  if      (n < 1024){ p=p0; g=32; s=8;  off=0;    }
  else if (n < 1280){ p=p1; g=16; s=16; off=1024; }
  else if (n < 1344){ p=p2; g=8;  s=32; off=1280; }
  else              { p=p3; g=4;  s=64; off=1344; }
  int hw = n - off, y = hw / g, x = hw % g;
  size_t cs = (size_t)g*g;
  size_t base = (size_t)b*8*cs + (size_t)y*g + x;
  float q0=p[base], q1=p[base+cs], q2=p[base+2*cs], q3=p[base+3*cs], q4=p[base+4*cs];
  float c5=p[base+5*cs], c6=p[base+6*cs], c7=p[base+7*cs];
  float cx=(q0+0.5f+(float)x)*(float)s, cy=(q1+0.5f+(float)y)*(float)s;
  float w=__expf(q2)*(float)s, h=__expf(q3)*(float)s;
  float* bo = boxes + (size_t)id*4;
  bo[0]=cx-0.5f*w; bo[1]=cy-0.5f*h; bo[2]=cx+0.5f*w; bo[3]=cy+0.5f*h;
  scores[id] = sigmf(q4);
  int lab = 0; float best = c5;
  if (c6 > best){ best=c6; lab=1; }
  if (c7 > best){ best=c7; lab=2; }
  labels[id] = lab;
}

// ---------------- bitonic sort (desc, idx tie-break) ----------------
__device__ void bitonic_desc(float* key, int* idx, int n, int tid){
  for (int k=2; k<=n; k<<=1){
    for (int j=k>>1; j>0; j>>=1){
      for (int i=tid; i<n; i+=256){
        int l = i ^ j;
        if (l > i){
          float ki=key[i], kl=key[l]; int ii=idx[i], il=idx[l];
          bool greater = (ki > kl) || (ki==kl && ii < il);
          bool descBlk = ((i & k) == 0);
          bool doswap  = descBlk ? !greater : greater;
          if (doswap){ key[i]=kl; key[l]=ki; idx[i]=il; idx[l]=ii; }
        }
      }
      __syncthreads();
    }
  }
}

// ---------------- per-image NMS ----------------
__global__ __launch_bounds__(256) void nms_kernel(
    const float* __restrict__ boxes, const float* __restrict__ scores,
    const int* __restrict__ labels,
    float* __restrict__ bsel, float* __restrict__ ssel,
    int* __restrict__ lsel, int* __restrict__ vsel)
{
  __shared__ float skey[2048];
  __shared__ int   sidx[2048];
  __shared__ float bxo[KPRE*4];   // class-offset boxes
  __shared__ float bpl[KPRE*4];   // plain boxes
  __shared__ float ssc[KPRE];
  __shared__ int   slb[KPRE];
  __shared__ int   skp[KPRE];
  const float NEG = -INFINITY;
  int b = blockIdx.x, tid = threadIdx.x;

  for (int i=tid; i<2048; i+=256){
    float k = NEG;
    if (i < NCAND){ float s = scores[b*NCAND+i]; if (s > 0.5f) k = s; }
    skey[i]=k; sidx[i]=i;
  }
  __syncthreads();
  bitonic_desc(skey, sidx, 2048, tid);

  for (int i=tid; i<KPRE; i+=256){
    int j = sidx[i]; float s = skey[i];
    int lab = labels[b*NCAND+j];
    const float* bo = boxes + ((size_t)(b*NCAND+j))*4;
    float offc = (float)lab * 1024.0f;       // 4 * PATCH
    bpl[i*4+0]=bo[0]; bpl[i*4+1]=bo[1]; bpl[i*4+2]=bo[2]; bpl[i*4+3]=bo[3];
    bxo[i*4+0]=bo[0]+offc; bxo[i*4+1]=bo[1]+offc; bxo[i*4+2]=bo[2]+offc; bxo[i*4+3]=bo[3]+offc;
    ssc[i]=s; slb[i]=lab; skp[i] = (s != NEG) ? 1 : 0;
  }
  __syncthreads();

  for (int i=0; i<KPRE; i++){
    if (skp[i]){
      float ax0=bxo[i*4], ay0=bxo[i*4+1], ax1=bxo[i*4+2], ay1=bxo[i*4+3];
      float aar = fmaxf(ax1-ax0,0.f)*fmaxf(ay1-ay0,0.f);
      for (int j2=i+1+tid; j2<KPRE; j2+=256){
        if (skp[j2]){
          float bx0=bxo[j2*4], by0=bxo[j2*4+1], bx1=bxo[j2*4+2], by1=bxo[j2*4+3];
          float bar = fmaxf(bx1-bx0,0.f)*fmaxf(by1-by0,0.f);
          float iw = fmaxf(fminf(ax1,bx1)-fmaxf(ax0,bx0), 0.f);
          float ih = fmaxf(fminf(ay1,by1)-fmaxf(ay0,by0), 0.f);
          float inter = iw*ih;
          float iou = inter / fmaxf(aar + bar - inter, 1e-9f);
          if (iou > 0.5f) skp[j2] = 0;
        }
      }
    }
    __syncthreads();
  }

  for (int i=tid; i<KPRE; i+=256){ skey[i] = skp[i] ? ssc[i] : NEG; sidx[i]=i; }
  __syncthreads();
  bitonic_desc(skey, sidx, KPRE, tid);

  for (int r=tid; r<RLIM; r+=256){
    float s = skey[r]; int q = sidx[r];
    int v = (s != NEG) ? 1 : 0;
    int o = b*RLIM + r;
    ssel[o] = v ? ssc[q] : 0.f;
    lsel[o] = v ? slb[q] : 0;
    vsel[o] = v;
    for (int c=0;c<4;c++) bsel[o*4+c] = v ? bpl[q*4+c] : 0.f;
  }
}

// ---------------- weight panel prep (WMMA B lane layout, f16) ----------------
__global__ __launch_bounds__(256) void prep_panel_kernel(
    const float* __restrict__ w1, _Float16* __restrict__ panel)
{
  int f = blockIdx.x*256 + threadIdx.x;
  if (f >= KDIM*64) return;
  int slot = f & 15;
  int lane = (f >> 4) & 31;
  int p    = f >> 9;         // panel id
  int nt   = p & 3;
  int jt   = p >> 2;
  int j    = jt % 9;         // ky*3+kx
  int t    = jt / 9;         // channel chunk
  int laneHi = lane >> 4;
  int n = nt*16 + (lane & 15);
  int klocal = (slot & 7) + (laneHi ? 8 : 0) + ((slot >> 3) ? 16 : 0);
  int c = t*32 + klocal;
  panel[f] = (_Float16)w1[(size_t)n*KDIM + (size_t)c*9 + j];
}

// ---------------- fused ROI-align + 3x3 conv (WMMA) + 1x1 conv + sigmoid ----------------
__global__ __launch_bounds__(256) void roi_conv_kernel(
    const float* __restrict__ mf0, const float* __restrict__ mf1,
    const float* __restrict__ mf2, const float* __restrict__ mf3,
    const float* __restrict__ bsel, const _Float16* __restrict__ panel,
    const float* __restrict__ b1,  const float* __restrict__ w2,
    const float* __restrict__ b2,  float* __restrict__ lm)
{
  // zero-padded halo tile: [18 rows][18 cols][32 ch] f16 = 10368 halves (20.25 KB)
  __shared__ _Float16 S[SHALVES];
  int roi = blockIdx.x;
  int tid = threadIdx.x;
  int b = roi / RLIM;
  int lane = tid & 31, wave = tid >> 5, laneHi = lane >> 4;

  // ---- zero entire tile once (halo stays zero; interior is overwritten each chunk) ----
  {
    v8h z;
    #pragma unroll
    for (int e=0; e<8; e++) z[e] = (_Float16)0.f;
    for (int i = tid; i < SHALVES/8; i += 256)
      *reinterpret_cast<v8h*>(&S[i*8]) = z;
  }

  // ---- per-position bilinear sampling setup (ROI align on 64x64 merged grid) ----
  float x0 = bsel[roi*4+0], y0 = bsel[roi*4+1], x1 = bsel[roi*4+2], y1 = bsel[roi*4+3];
  float rw = fmaxf(x1-x0, 1.f), rh = fmaxf(y1-y0, 1.f);
  int ppy = tid >> 4, ppx = tid & 15;
  float xs = x0 + ((float)ppx + 0.5f) * rw * 0.0625f;
  float ys = y0 + ((float)ppy + 0.5f) * rh * 0.0625f;
  bool oob = (ys < -1.f) || (ys > 64.f) || (xs < -1.f) || (xs > 64.f);
  float yc = fminf(fmaxf(ys, 0.f), 63.f);
  float xc = fminf(fmaxf(xs, 0.f), 63.f);
  int y0i = (int)floorf(yc), x0i = (int)floorf(xc);
  int y1i = min(y0i+1, 63),  x1i = min(x0i+1, 63);
  float ly = yc - (float)y0i, lx = xc - (float)x0i;
  float w00=(1.f-ly)*(1.f-lx), w01=(1.f-ly)*lx, w10=ly*(1.f-lx), w11=ly*lx;
  if (oob){ w00=w01=w10=w11=0.f; }

  // interior staging base for this thread's position
  int sbase = ((ppy+1)*PADW + (ppx+1))*32;
  // A-fragment column base for this lane (row added per (mt,dy))
  int colbase = ((lane & 15) + 1)*32 + (laneHi ? 8 : 0);

  float b1v[4], w2v[4];
  #pragma unroll
  for (int nt=0; nt<4; nt++){ int n = nt*16 + (lane & 15); b1v[nt]=b1[n]; w2v[nt]=w2[n]; }
  float b2v = b2[0];

  v8f acc[2][4];
  #pragma unroll
  for (int mt=0; mt<2; mt++)
    #pragma unroll
    for (int nt=0; nt<4; nt++)
      #pragma unroll
      for (int q=0; q<8; q++) acc[mt][nt][q] = 0.f;

  const float* mfs[4] = {mf0, mf1, mf2, mf3};
  const int ClA[4]={32,64,128,256}, CstA[4]={0,32,96,224}, GsA[4]={64,32,16,8};

  __syncthreads();  // halo zero visible before first A reads

  for (int t=0; t<NCHUNK; t++){
    int cbase = t*32;
    int lvl = (cbase<32)?0 : (cbase<96)?1 : (cbase<224)?2 : 3;
    int g = GsA[lvl], sh = lvl;
    int gy0=(y0i>>sh)*g, gy1=(y1i>>sh)*g, gx0=(x0i>>sh), gx1=(x1i>>sh);
    int a00=gy0+gx0, a01=gy0+gx1, a10=gy1+gx0, a11=gy1+gx1;
    const float* basep = mfs[lvl] + ((size_t)b*ClA[lvl] + (size_t)(cbase - CstA[lvl])) * (size_t)(g*g);

    __syncthreads();  // previous chunk fully consumed
    #pragma unroll
    for (int q=0; q<4; q++){
      v8h tmp;
      #pragma unroll
      for (int e=0; e<8; e++){
        const float* pch = basep + (size_t)(q*8+e) * (size_t)(g*g);
        float v = pch[a00]*w00 + pch[a01]*w01 + pch[a10]*w10 + pch[a11]*w11;
        tmp[e] = (_Float16)v;
      }
      *reinterpret_cast<v8h*>(&S[sbase + q*8]) = tmp;
    }
    __syncthreads();

    const _Float16* pj = panel + (size_t)t*9*4*512;
    #pragma unroll 3
    for (int j=0; j<9; j++){
      int dy = j/3 - 1, dx = j%3 - 1;
      v16h bf[4];
      #pragma unroll
      for (int nt=0; nt<4; nt++)
        bf[nt] = *reinterpret_cast<const v16h*>(pj + ((size_t)j*4 + nt)*512 + lane*16);
      #pragma unroll
      for (int mt=0; mt<2; mt++){
        int py = 2*wave + mt;
        int base = (py + dy + 1)*(PADW*32) + colbase + dx*32;  // always in-bounds (halo)
        v8h lo = *reinterpret_cast<const v8h*>(&S[base]);
        v8h hi = *reinterpret_cast<const v8h*>(&S[base + 16]);
        v16h a = __builtin_shufflevector(lo, hi, 0,1,2,3,4,5,6,7,8,9,10,11,12,13,14,15);
        #pragma unroll
        for (int nt=0; nt<4; nt++)
          acc[mt][nt] = __builtin_amdgcn_wmma_f32_16x16x32_f16(
              false, a, false, bf[nt], (short)0, acc[mt][nt], false, false);
      }
    }
  }

  // ---- epilogue: relu(acc+b1)*w2 summed over 64 channels, +b2, sigmoid ----
  #pragma unroll
  for (int mt=0; mt<2; mt++){
    int py = 2*wave + mt;
    #pragma unroll
    for (int i=0; i<8; i++){
      float s = 0.f;
      #pragma unroll
      for (int nt=0; nt<4; nt++)
        s += fmaxf(acc[mt][nt][i] + b1v[nt], 0.f) * w2v[nt];
      s += __shfl_xor(s, 1);
      s += __shfl_xor(s, 2);
      s += __shfl_xor(s, 4);
      s += __shfl_xor(s, 8);
      s = sigmf(s + b2v);
      if ((lane & 15) == 0){
        int px = i + (laneHi ? 8 : 0);
        lm[(size_t)roi*256 + py*16 + px] = s;
      }
    }
  }
}

// ---------------- counts init ----------------
__global__ void init_counts_kernel(int* __restrict__ counts){
  int i = blockIdx.x*256 + threadIdx.x;
  if (i < NROI) counts[i] = 0;
}

// ---------------- paste masks ----------------
__global__ __launch_bounds__(256) void paste_kernel(
    const float* __restrict__ bsel, const float* __restrict__ lm,
    float* __restrict__ mask_out, int* __restrict__ counts)
{
  int roi = blockIdx.y, y = blockIdx.x, x = threadIdx.x;
  float bf0=bsel[roi*4+0], bf1=bsel[roi*4+1], bf2=bsel[roi*4+2], bf3=bsel[roi*4+3];
  int bx0 = min(max((int)bf0, 0), PATCHSZ-1);
  int by0 = min(max((int)bf1, 0), PATCHSZ-1);
  int bx1 = min(max((int)bf2, 0), PATCHSZ-1);
  int by1 = min(max((int)bf3, 0), PATCHSZ-1);
  int h = by1 - by0 + 1, w = bx1 - bx0 + 1;
  bool inside = (y>=by0) && (y<=by1) && (x>=bx0) && (x<=bx1);
  int m = 0;
  if (inside){
    int sy = min(((y-by0)*MSZ)/h, MSZ-1);
    int sx = min(((x-bx0)*MSZ)/w, MSZ-1);
    m = (lm[roi*256 + sy*16 + sx] > 0.5f) ? 1 : 0;
  }
  mask_out[(size_t)roi*65536 + (size_t)y*256 + x] = m ? 1.f : 0.f;

  __shared__ int cnt;
  if (threadIdx.x == 0) cnt = 0;
  __syncthreads();
  unsigned long long bal = __ballot(m != 0);
  if ((threadIdx.x & 31) == 0 && bal) atomicAdd(&cnt, __popcll(bal));
  __syncthreads();
  if (threadIdx.x == 0 && cnt > 0) atomicAdd(&counts[roi], cnt);
}

// ---------------- finalize ----------------
__global__ __launch_bounds__(256) void finalize_kernel(
    const float* __restrict__ bsel, const float* __restrict__ ssel,
    const int* __restrict__ lsel, const int* __restrict__ vsel,
    const int* __restrict__ counts,
    float* __restrict__ out_scores, float* __restrict__ out_boxes,
    float* __restrict__ out_labels, float* __restrict__ out_masks,
    float* __restrict__ out_vout)
{
  int r = blockIdx.x;
  __shared__ int vsh;
  if (threadIdx.x == 0){
    int v = (vsel[r] != 0) && (counts[r] > 0);
    out_scores[r] = v ? ssel[r] : 0.f;
    out_labels[r] = v ? (float)lsel[r] : 0.f;
    out_vout[r]   = v ? 1.f : 0.f;
    for (int c=0;c<4;c++) out_boxes[r*4+c] = v ? bsel[r*4+c] : 0.f;
    vsh = v;
  }
  __syncthreads();
  if (!vsh){
    float* mp = out_masks + (size_t)r*65536;
    for (int i=threadIdx.x; i<65536; i+=256) mp[i] = 0.f;
  }
}

// ---------------- launch ----------------
extern "C" void kernel_launch(void* const* d_in, const int* in_sizes, int n_in,
                              void* d_out, int out_size, void* d_ws, size_t ws_size,
                              hipStream_t stream)
{
  (void)in_sizes; (void)n_in; (void)out_size; (void)ws_size;
  const float* p0  = (const float*)d_in[0];
  const float* p1  = (const float*)d_in[1];
  const float* p2  = (const float*)d_in[2];
  const float* p3  = (const float*)d_in[3];
  const float* mf0 = (const float*)d_in[4];
  const float* mf1 = (const float*)d_in[5];
  const float* mf2 = (const float*)d_in[6];
  const float* mf3 = (const float*)d_in[7];
  const float* w1  = (const float*)d_in[8];
  const float* b1  = (const float*)d_in[9];
  const float* w2  = (const float*)d_in[10];
  const float* b2  = (const float*)d_in[11];

  char* ws = (char*)d_ws;
  float*     boxes_all  = (float*)(ws + OFF_BOXES);
  float*     scores_all = (float*)(ws + OFF_SCORES);
  int*       labels_all = (int*)  (ws + OFF_LABELS);
  float*     bsel       = (float*)(ws + OFF_BSEL);
  float*     ssel       = (float*)(ws + OFF_SSEL);
  int*       lsel       = (int*)  (ws + OFF_LSEL);
  int*       vsel       = (int*)  (ws + OFF_VSEL);
  _Float16*  panel      = (_Float16*)(ws + OFF_PANEL);
  float*     lm         = (float*)(ws + OFF_LM);
  int*       counts     = (int*)  (ws + OFF_CNT);

  float* out        = (float*)d_out;
  float* out_scores = out;                       // [8,100]
  float* out_boxes  = out + NROI;                // [8,100,4]
  float* out_labels = out + NROI*5;              // [8,100]
  float* out_masks  = out + NROI*6;              // [8,100,256,256]
  float* out_vout   = out + NROI*6 + (size_t)NROI*65536; // [8,100]

  init_counts_kernel<<<(NROI+255)/256, 256, 0, stream>>>(counts);
  decode_kernel<<<(NB*NCAND+255)/256, 256, 0, stream>>>(p0,p1,p2,p3, boxes_all, scores_all, labels_all);
  nms_kernel<<<NB, 256, 0, stream>>>(boxes_all, scores_all, labels_all, bsel, ssel, lsel, vsel);
  prep_panel_kernel<<<(KDIM*64+255)/256, 256, 0, stream>>>(w1, panel);
  roi_conv_kernel<<<NROI, 256, 0, stream>>>(mf0,mf1,mf2,mf3, bsel, panel, b1, w2, b2, lm);
  paste_kernel<<<dim3(PATCHSZ, NROI), 256, 0, stream>>>(bsel, lm, out_masks, counts);
  finalize_kernel<<<NROI, 256, 0, stream>>>(bsel, ssel, lsel, vsel, counts,
                                            out_scores, out_boxes, out_labels, out_masks, out_vout);
}
